// MultiHeadGAT_27719718928466
// MI455X (gfx1250) — matrix operand order repaired and checked by reading, not compiled, not run on hardware
//
#include <hip/hip_runtime.h>
#include <hip/hip_bf16.h>

typedef __attribute__((ext_vector_type(16))) _Float16 v16h;
typedef __attribute__((ext_vector_type(8)))  float    v8f;

#define NHEAD 4
#define TD    768
#define HID   256
#define NB    16      // batch
#define LL    1024    // tokens
#define MTOT  (NB*LL) // 16384 rows
#define EPSN  1e-5f
#define SLOPE 0.01f

__device__ __forceinline__ float leaky(float e) { return e >= 0.f ? e : SLOPE * e; }

// ---- WMMA fragment gathers per CDNA5 ISA 7.12.2 (wave32) -------------------
// A (16x32 f16): lane 0-15 -> M=lane; VGPR0-3: K=(hi?8:0)+0..7, VGPR4-7: K=16+(hi?8:0)+0..7
__device__ __forceinline__ v16h frag_a_lds(const _Float16* t, int row_base, int stride) {
  const int lane = threadIdx.x & 31;
  const int m = row_base + (lane & 15);
  const int hi = lane >> 4;
  union { v16h v; _Float16 e[16]; } u;
#pragma unroll
  for (int r = 0; r < 8; ++r) {
    const int kk = ((r >= 4) ? 16 : 0) + hi * 8 + (r & 3) * 2;
    u.e[2*r]   = t[m*stride + kk];
    u.e[2*r+1] = t[m*stride + kk + 1];
  }
  return u.v;
}
// B (32x16 f16): N = lane%16; VGPR0-3: K=(hi?8:0)+0..7, VGPR4-7: K=16+(hi?8:0)+0..7 (2/VGPR).
// LDS tile is stored transposed [N][K] so each f16 pair is one b32 load.
__device__ __forceinline__ v16h frag_b_ldsT(const _Float16* t, int col_base, int stride) {
  const int lane = threadIdx.x & 31;
  const int n = col_base + (lane & 15);
  const int hi = lane >> 4;
  union { v16h v; _Float16 e[16]; } u;
#pragma unroll
  for (int r = 0; r < 8; ++r) {
    const int kk = ((r >= 4) ? 16 : 0) + hi * 8 + (r & 3) * 2;
    u.e[2*r]   = t[n*stride + kk];
    u.e[2*r+1] = t[n*stride + kk + 1];
  }
  return u.v;
}

// ---------------------------------------------------------------------------
// Kernel 1: per-head projection GEMM + bias + LayerNorm + attention dot pre-
// computation.  grid = (256 m-tiles, 4 heads), block = 256 (8 waves).
// Each workgroup computes a 64x256 tile of h for one head.
// ---------------------------------------------------------------------------
__global__ __launch_bounds__(256) void k_proj_ln(
    const float* __restrict__ x,     // [MTOT, 768]
    const float* __restrict__ Wfc,   // [4, 768, 256]
    const float* __restrict__ bfc,   // [4, 256]
    const float* __restrict__ lng,   // [4, 256]
    const float* __restrict__ lnb,   // [4, 256]
    const float* __restrict__ aw,    // [4, 512]
    _Float16* __restrict__ hout,     // [4, MTOT, 256] f16
    float* __restrict__ srow,        // [4, MTOT]
    float* __restrict__ scol)        // [4, MTOT]
{
  __shared__ _Float16 At[64*40];     // A tile 64x32 (stride 40 halves)
  __shared__ _Float16 Bt[HID*40];    // B tile transposed [n][k] 256x32
  __shared__ float hbuf[64*HID];     // f32 result tile for LN
  __shared__ float p_bfc[HID], p_g[HID], p_b[HID], p_a1[HID], p_a2[HID];
  __shared__ float mu_s[64], rs_s[64], ssr[64], ssc[64];

  const int tid  = threadIdx.x;
  const int head = blockIdx.y;
  const int m0   = blockIdx.x * 64;

  p_bfc[tid] = bfc[head*HID + tid];
  p_g[tid]   = lng[head*HID + tid];
  p_b[tid]   = lnb[head*HID + tid];
  p_a1[tid]  = aw[head*2*HID + tid];
  p_a2[tid]  = aw[head*2*HID + HID + tid];

  const int w = tid >> 5, lane = tid & 31, hi = lane >> 4;
  const int rb = (w & 3) * 16;       // wave row block (16 rows)
  const int c0 = (w >> 2) * 128;     // wave col half (8 WMMA tiles)

  v8f zero = {};
  v8f acc[8];
#pragma unroll
  for (int t = 0; t < 8; ++t) acc[t] = zero;

  for (int kc = 0; kc < TD; kc += 32) {
    { // stage A: 2048 f32 -> f16, 8 contiguous per thread
      const int r = tid >> 2;
      const int c = (tid & 3) * 8;
      const float* gp = x + (size_t)(m0 + r) * TD + kc + c;
#pragma unroll
      for (int u = 0; u < 8; ++u) At[r*40 + c + u] = (_Float16)gp[u];
    }
    { // stage B transposed: 8192 f32 -> f16
      const int k  = tid >> 3;
      const int nb = (tid & 7) * 32;
      const float* gp = Wfc + ((size_t)head*TD + kc + k) * HID + nb;
#pragma unroll
      for (int u = 0; u < 32; ++u) Bt[(nb + u)*40 + k] = (_Float16)gp[u];
    }
    __syncthreads();
    v16h af = frag_a_lds(At, rb, 40);
#pragma unroll
    for (int t = 0; t < 8; ++t) {
      v16h bf = frag_b_ldsT(Bt, c0 + t*16, 40);
      acc[t] = __builtin_amdgcn_wmma_f32_16x16x32_f16(
          false, af, false, bf, (short)0, acc[t], false, false);
    }
    __syncthreads();
  }

  // write accumulators (+bias) to LDS; C/D layout: row = v + 8*hi, col = lane%16
#pragma unroll
  for (int t = 0; t < 8; ++t) {
    const int n = c0 + t*16 + (lane & 15);
#pragma unroll
    for (int v = 0; v < 8; ++v) {
      const int r = rb + v + hi*8;
      hbuf[r*HID + n] = acc[t][v] + p_bfc[n];
    }
  }
  __syncthreads();

  if (tid < 64) {  // LN stats per row
    float s = 0.f, s2 = 0.f;
    for (int c = 0; c < HID; ++c) { float xv = hbuf[tid*HID + c]; s += xv; s2 += xv*xv; }
    float mu = s * (1.f/HID);
    float var = s2 * (1.f/HID) - mu*mu;
    mu_s[tid] = mu;
    rs_s[tid] = rsqrtf(var + EPSN);
    ssr[tid] = 0.f; ssc[tid] = 0.f;
  }
  __syncthreads();

  { // normalize, emit f16 h, partial attention dots (4 threads/row)
    const int r  = tid >> 2;
    const int cq = (tid & 3) * 64;
    const float mu = mu_s[r], rs = rs_s[r];
    float s1 = 0.f, s2v = 0.f;
    _Float16* op = hout + ((size_t)head*MTOT + m0 + r) * HID + cq;
    for (int c = 0; c < 64; ++c) {
      const int cc = cq + c;
      float hn = (hbuf[r*HID + cc] - mu) * rs * p_g[cc] + p_b[cc];
      op[c] = (_Float16)hn;
      s1  += hn * p_a1[cc];
      s2v += hn * p_a2[cc];
    }
    atomicAdd(&ssr[r], s1);   // ds_add_f32
    atomicAdd(&ssc[r], s2v);
  }
  __syncthreads();
  if (tid < 64) {
    srow[head*MTOT + m0 + tid] = ssr[tid];
    scol[head*MTOT + m0 + tid] = ssc[tid];
  }
}

// ---------------------------------------------------------------------------
// Kernel 2: softmax(leaky(s_col[i]+s_row[j]+ab)) @ h, fused with mean over i.
// h_prime never hits memory; token-mean accumulated into sent_acc with atomics.
// grid = (16 i-tiles, 16 batches, 4 heads), block = 256 (8 waves, 64x256 tile).
// ---------------------------------------------------------------------------
__global__ __launch_bounds__(256) void k_attn(
    const _Float16* __restrict__ h,   // [4, MTOT, 256] f16
    const float* __restrict__ srow,
    const float* __restrict__ scol,
    const float* __restrict__ ab,     // [4]
    float* __restrict__ sent_acc)     // [16, 1024] (pre-zeroed)
{
  __shared__ _Float16 Pt[64*40];      // P tile (exp weights) 64x32
  __shared__ _Float16 Ht[HID*40];     // h chunk transposed [n][k] 256x32
  __shared__ float sc_s[64], mx_s[64], den_s[64];
  __shared__ float red[256];
  __shared__ float sr_s[32];
  __shared__ float Mr_s;

  const int tid  = threadIdx.x;
  const int it   = blockIdx.x;
  const int b    = blockIdx.y;
  const int head = blockIdx.z;
  const size_t base = (size_t)head*MTOT + (size_t)b*LL;
  const float abv = ab[head];

  { // Mr = max_j s_row  (leaky is monotonic -> row max in O(L))
    float m = -1e30f;
    for (int j = tid; j < LL; j += 256) m = fmaxf(m, srow[base + j]);
    red[tid] = m;
    __syncthreads();
    for (int s = 128; s > 0; s >>= 1) {
      if (tid < s) red[tid] = fmaxf(red[tid], red[tid + s]);
      __syncthreads();
    }
    if (tid == 0) Mr_s = red[0];
    __syncthreads();
  }
  if (tid < 64) {
    float sc = scol[base + it*64 + tid];
    sc_s[tid]  = sc;
    mx_s[tid]  = leaky(sc + Mr_s + abv);
    den_s[tid] = 0.f;
  }

  const int w = tid >> 5, lane = tid & 31, hi = lane >> 4;
  const int rb = (w & 3) * 16;
  const int c0 = (w >> 2) * 128;
  v8f zero = {};
  v8f acc[8];
#pragma unroll
  for (int t = 0; t < 8; ++t) acc[t] = zero;

  for (int jc = 0; jc < LL; jc += 32) {
    if (tid < 32) sr_s[tid] = srow[base + jc + tid];
    __syncthreads();
    { // stage h chunk transposed (uint4 global reads, b16 LDS scatter)
      const int k  = tid >> 3;
      const int nb = (tid & 7) * 32;
      const uint4* gp = (const uint4*)(h + (base + jc + k) * HID + nb);
      union { uint4 q; _Float16 e[8]; } u;
#pragma unroll
      for (int q4 = 0; q4 < 4; ++q4) {
        u.q = gp[q4];
#pragma unroll
        for (int e8 = 0; e8 < 8; ++e8) Ht[(nb + q4*8 + e8)*40 + k] = u.e[e8];
      }
    }
    { // P tile + running denominator (ds_add_f32)
      const int r  = tid & 63;
      const int j0 = (tid >> 6) * 8;
      const float scr = sc_s[r], mxr = mx_s[r];
      float psum = 0.f;
#pragma unroll
      for (int u = 0; u < 8; ++u) {
        float e = leaky(scr + sr_s[j0 + u] + abv);
        float p = __expf(e - mxr);     // <= 1, f16-safe
        psum += p;
        Pt[r*40 + j0 + u] = (_Float16)p;
      }
      atomicAdd(&den_s[r], psum);
    }
    __syncthreads();
    v16h af = frag_a_lds(Pt, rb, 40);
#pragma unroll
    for (int t = 0; t < 8; ++t) {
      v16h bf = frag_b_ldsT(Ht, c0 + t*16, 40);
      acc[t] = __builtin_amdgcn_wmma_f32_16x16x32_f16(
          false, af, false, bf, (short)0, acc[t], false, false);
    }
    __syncthreads();
  }

  // normalize rows and fold the 16-row tile into per-column sums -> sent_acc
#pragma unroll
  for (int t = 0; t < 8; ++t) {
    const int n = c0 + t*16 + (lane & 15);
    float cs = 0.f;
#pragma unroll
    for (int v = 0; v < 8; ++v) {
      const int r = rb + v + hi*8;
      cs += acc[t][v] / den_s[r];
    }
    cs += __shfl_xor(cs, 16, 32);      // fold lane halves (row halves)
    if (lane < 16)
      atomicAdd(&sent_acc[b*(NHEAD*HID) + head*HID + n], cs);
  }
}

// ---------------------------------------------------------------------------
// Tail kernels (tiny, f32 scalar)
// ---------------------------------------------------------------------------
__global__ void k_zero(float* __restrict__ p, int n) {
  int i = blockIdx.x * 256 + threadIdx.x;
  if (i < n) p[i] = 0.f;
}

__global__ __launch_bounds__(256) void k_sent_mm(
    const float* __restrict__ sacc, const float* __restrict__ Wcat,
    const float* __restrict__ bcat, float* __restrict__ z)
{
  const int o = blockIdx.x * 256 + threadIdx.x;  // 0..12287
  const int b = o / 768, c = o % 768;
  const float* sp = sacc + b * 1024;
  float s = bcat[c];
  for (int k = 0; k < 1024; ++k)
    s += (sp[k] * (1.f / (float)LL)) * Wcat[k*768 + c];
  z[o] = s;
}

__global__ __launch_bounds__(256) void k_ln_out(
    const float* __restrict__ z, const float* __restrict__ g,
    const float* __restrict__ bb, float* __restrict__ sent_ln,
    float* __restrict__ out_sent)
{
  __shared__ float red[256], red2[256];
  __shared__ float mu_s, rs_s;
  const int row = blockIdx.x, tid = threadIdx.x;
  float s = 0.f, s2 = 0.f;
  for (int c = tid; c < 768; c += 256) { float v = z[row*768 + c]; s += v; s2 += v*v; }
  red[tid] = s; red2[tid] = s2;
  __syncthreads();
  for (int st = 128; st > 0; st >>= 1) {
    if (tid < st) { red[tid] += red[tid+st]; red2[tid] += red2[tid+st]; }
    __syncthreads();
  }
  if (tid == 0) {
    float mu = red[0] / 768.f;
    mu_s = mu;
    rs_s = rsqrtf(red2[0] / 768.f - mu*mu + EPSN);
  }
  __syncthreads();
  for (int c = tid; c < 768; c += 256) {
    float v = (z[row*768 + c] - mu_s) * rs_s * g[c] + bb[c];
    sent_ln[row*768 + c] = v;
    out_sent[row*768 + c] = v;
  }
}

__global__ __launch_bounds__(256) void k_dims(
    const float* __restrict__ sln, const float* __restrict__ Wfcs,
    const float* __restrict__ bfcs, float* __restrict__ out)
{
  const int idx = blockIdx.x * 256 + threadIdx.x;  // < 36864
  const int i = idx / 12288, rem = idx % 12288;
  const int b = rem / 768, c = rem % 768;
  const float* sp = sln + b * 768;
  const float* wp = Wfcs + (size_t)i * 768 * 768 + c;
  float s = bfcs[i*768 + c];
  for (int k = 0; k < 768; ++k) s += sp[k] * wp[k*768];
  out[idx] = s;
}

// ---------------------------------------------------------------------------
extern "C" void kernel_launch(void* const* d_in, const int* in_sizes, int n_in,
                              void* d_out, int out_size, void* d_ws, size_t ws_size,
                              hipStream_t stream) {
  const float* x    = (const float*)d_in[0];
  const float* Wfc  = (const float*)d_in[1];
  const float* bfc  = (const float*)d_in[2];
  const float* lng  = (const float*)d_in[3];
  const float* lnb  = (const float*)d_in[4];
  const float* aw   = (const float*)d_in[5];
  const float* ab   = (const float*)d_in[6];
  const float* Wcat = (const float*)d_in[7];
  const float* bcat = (const float*)d_in[8];
  const float* lOg  = (const float*)d_in[9];
  const float* lOb  = (const float*)d_in[10];
  const float* Wfcs = (const float*)d_in[11];
  const float* bfcs = (const float*)d_in[12];
  float* out = (float*)d_out;

  char* ws = (char*)d_ws;
  size_t off = 0;
  auto alloc = [&](size_t bytes) -> char* {
    char* p = ws + off;
    off = (off + bytes + 255) & ~(size_t)255;
    return p;
  };
  _Float16* hbuf = (_Float16*)alloc((size_t)NHEAD * MTOT * HID * 2);  // 32 MB
  float* srow = (float*)alloc((size_t)NHEAD * MTOT * 4);
  float* scol = (float*)alloc((size_t)NHEAD * MTOT * 4);
  float* sacc = (float*)alloc((size_t)NB * 1024 * 4);
  float* z    = (float*)alloc((size_t)NB * 768 * 4);
  float* sln  = (float*)alloc((size_t)NB * 768 * 4);
  (void)ws_size; (void)in_sizes; (void)n_in; (void)out_size;

  k_zero<<<dim3(64), dim3(256), 0, stream>>>(sacc, NB * 1024);
  k_proj_ln<<<dim3(MTOT/64, NHEAD), dim3(256), 0, stream>>>(
      x, Wfc, bfc, lng, lnb, aw, hbuf, srow, scol);
  k_attn<<<dim3(LL/64, NB, NHEAD), dim3(256), 0, stream>>>(
      hbuf, srow, scol, ab, sacc);
  k_sent_mm<<<dim3(48), dim3(256), 0, stream>>>(sacc, Wcat, bcat, z);
  k_ln_out<<<dim3(NB), dim3(256), 0, stream>>>(z, lOg, lOb, sln, out + 3*12288);
  k_dims<<<dim3(144), dim3(256), 0, stream>>>(sln, Wfcs, bfcs, out);
}
